// AgentAwareAttention_84043920048563
// MI455X (gfx1250) — matrix operand order repaired and checked by reading, not compiled, not run on hardware
//
#include <hip/hip_runtime.h>

// ---------------- problem constants ----------------
#define B2     2
#define NN     2009          // real sequence length
#define NP     2016          // padded to 126 * 16
#define NT     (NP/16)       // 126 row/col tiles
#define DIMM   512
#define HEADS  8
#define DH     64
#define SCALE  0.125f        // 64^-0.5
#define PCOLS  2560          // 1536 (qkv) + 1024 (q_self,k_self)
#define AGENT_LIM 2000       // NUM_AGENT * IN_LEN

typedef __attribute__((ext_vector_type(16))) _Float16 v16h;
typedef __attribute__((ext_vector_type(8)))  _Float16 v8h;
typedef __attribute__((ext_vector_type(4)))  _Float16 v4h;
typedef __attribute__((ext_vector_type(8)))  float    v8f;
typedef __attribute__((ext_vector_type(4)))  float    f4;
typedef float f4u __attribute__((ext_vector_type(4), aligned(4)));  // unaligned-ok store

// WMMA f16 fragment maps (CDNA5 ISA 7.12.2, wave32):
//  A (16x32): row M = lane%16; lane halves = K[k0+8*khi .. +8) ++ K[k0+16+8*khi .. +8)
//  B (32x16): col N = lane%16; lane halves = K[k0+8*khi .. +8) ++ K[k0+16+8*khi .. +8)
//  C/D (16x16 f32): elem r -> row M = r + 8*(lane>=16), col N = lane%16

__device__ __forceinline__ v16h cat16(v8h lo, v8h hi) {
    return __builtin_shufflevector(lo, hi, 0,1,2,3,4,5,6,7,8,9,10,11,12,13,14,15);
}
__device__ __forceinline__ v16h load_afrag(const _Float16* rowptr, int k0, int khi) {
    v8h lo = *(const v8h*)(rowptr + k0 + khi * 8);
    v8h hi = *(const v8h*)(rowptr + k0 + 16 + khi * 8);
    return cat16(lo, hi);
}
__device__ __forceinline__ v16h load_bfrag(const _Float16* rowptr, int k0, int khi) {
    v8h lo = *(const v8h*)(rowptr + k0 + khi * 8);
    v8h hi = *(const v8h*)(rowptr + k0 + 16 + khi * 8);
    return cat16(lo, hi);
}

// ---------------- kernel 1: fp32 -> f16 staging -----------------------------
// x padded to NP rows (vectorized 8/thread); weights stored TRANSPOSED
// ([out_col][in_dim]) so GEMM B-fragments are K-contiguous per lane.
__global__ void __launch_bounds__(256)
k_aaa_convert(const float* __restrict__ x, const float* __restrict__ w_qkv,
              const float* __restrict__ w_self, const float* __restrict__ w_out,
              _Float16* __restrict__ xh, _Float16* __restrict__ whT,
              _Float16* __restrict__ wohT) {
    size_t i = (size_t)blockIdx.x * 256 + threadIdx.x;

    const size_t nxh8 = (size_t)B2 * NP * DIMM / 8;     // 8 halves per thread
    if (i < nxh8) {
        size_t e0 = i * 8;
        int d = (int)(e0 % DIMM);
        size_t nd = e0 / DIMM;
        int n = (int)(nd % NP);
        int b = (int)(nd / NP);
        v8h hv;
        if (n < NN) {
            const float* src = x + ((size_t)b * NN + n) * DIMM + d;
            f4 a = *(const f4*)(src);
            f4 c = *(const f4*)(src + 4);
#pragma unroll
            for (int t = 0; t < 4; ++t) { hv[t] = (_Float16)a[t]; hv[4 + t] = (_Float16)c[t]; }
        } else {
            hv = (v8h)(_Float16)0.0f;
        }
        *(v8h*)(xh + e0) = hv;
    }
    if (i < (size_t)PCOLS * DIMM) {          // whT[c][d] = W[d][c]
        int d = (int)(i % DIMM);
        int c = (int)(i / DIMM);
        float v = (c < 1536) ? w_qkv[(size_t)d * 1536 + c]
                             : w_self[(size_t)d * 1024 + (c - 1536)];
        whT[i] = (_Float16)v;
    }
    if (i < (size_t)DIMM * DIMM) {           // wohT[c][d] = w_out[d][c]
        int d = (int)(i % DIMM);
        int c = (int)(i / DIMM);
        wohT[i] = (_Float16)w_out[(size_t)d * DIMM + c];
    }
}

// ---------------- kernel 2: fused QKV + self projections --------------------
// P[b,n,c] for c in {Q:0..511, K:512..1023, Qs:1536..2047, Ks:2048..2559};
// V columns (1024..1535) are routed to a transposed buffer Vt[b,h,d,n].
__global__ void __launch_bounds__(256)
k_aaa_proj(const _Float16* __restrict__ xh, const _Float16* __restrict__ whT,
           _Float16* __restrict__ P, _Float16* __restrict__ Vt) {
    int wave = (int)((blockIdx.x * 256 + threadIdx.x) >> 5);
    int lane = threadIdx.x & 31;
    const int CT = PCOLS / 16;                 // 160
    int ct = wave % CT;
    int rest = wave / CT;
    int mt = rest % NT;
    int b  = rest / NT;
    if (b >= B2) return;

    int lrow = lane & 15;
    int khi  = (lane >= 16) ? 1 : 0;
    const _Float16* Arow = xh + ((size_t)b * NP + (size_t)mt * 16 + lrow) * DIMM;
    const _Float16* Brow = whT + ((size_t)ct * 16 + lrow) * DIMM;

    v8f acc = {};
#pragma unroll 4
    for (int k0 = 0; k0 < DIMM; k0 += 32) {
        v16h a  = load_afrag(Arow, k0, khi);
        v16h bb = load_bfrag(Brow, k0, khi);
        acc = __builtin_amdgcn_wmma_f32_16x16x32_f16(false, a, false, bb,
                                                     (short)0, acc, false, false);
    }

    bool isV = (ct >= 64) && (ct < 96);        // columns 1024..1535
    if (isV) {
        int col = ct * 16 + lrow - 1024;
        int h   = col >> 6;
        int d   = col & 63;
        int n0  = mt * 16 + khi * 8;
        v8h hv;
#pragma unroll
        for (int r = 0; r < 8; ++r) hv[r] = (_Float16)acc[r];
        *(v8h*)(Vt + (((size_t)b * HEADS + h) * DH + d) * NP + n0) = hv;
    } else {
        _Float16* Cp = P + ((size_t)b * NP + (size_t)mt * 16) * PCOLS + ct * 16;
#pragma unroll
        for (int r = 0; r < 8; ++r) {
            int row = r + khi * 8;
            Cp[(size_t)row * PCOLS + lrow] = (_Float16)acc[r];
        }
    }
}

// ---------------- kernel 3: fused attention ---------------------------------
// One block = (b, h, 16-row tile); 4 waves (128 thr) so phase-1 (~63 WMMA/wave)
// and phase-3 (63 WMMA/wave) are balanced with zero idle waves.
// Score/prob panel in LDS (f16, 63 KB) -> 5 blocks per 320KB WGP.
__global__ void __launch_bounds__(128)
k_aaa_attn(const _Float16* __restrict__ P, const _Float16* __restrict__ Vt,
           float* __restrict__ attn_out, _Float16* __restrict__ O) {
    __shared__ _Float16 panel[16 * NP];       // 64512 bytes

    int it = blockIdx.x % NT;
    int h  = (blockIdx.x / NT) % HEADS;
    int b  = blockIdx.x / (NT * HEADS);

    int wave = threadIdx.x >> 5;              // 0..3
    int lane = threadIdx.x & 31;
    int lrow = lane & 15;
    int khi  = (lane >= 16) ? 1 : 0;

    const _Float16* base  = P + (size_t)b * NP * PCOLS;
    const _Float16* Qrow  = base + (size_t)h * DH  + (size_t)(it * 16 + lrow) * PCOLS;
    const _Float16* Qsrow = base + 1536 + (size_t)h * DH + (size_t)(it * 16 + lrow) * PCOLS;
    const _Float16* Kbase = base + 512  + (size_t)h * DH;
    const _Float16* Ksbase= base + 2048 + (size_t)h * DH;

    // hoist Q / Qs fragments (invariant over column tiles)
    v16h aq0 = load_afrag(Qrow, 0, khi),  aq1 = load_afrag(Qrow, 32, khi);
    v16h as0 = load_afrag(Qsrow, 0, khi), as1 = load_afrag(Qsrow, 32, khi);

    // agent-diagonal band of this row tile
    int r0 = it * 16;
    int r1 = it * 16 + 15;
    int rr1 = (r1 < AGENT_LIM - 1) ? r1 : (AGENT_LIM - 1);
    int band_c0 = (r0 / 10) * 10;
    int band_c1 = (rr1 / 10) * 10 + 9;
    bool row_has_agents = (r0 < AGENT_LIM);

    // ---- phase 1: scores -> LDS panel ----
    for (int jt = wave; jt < NT; jt += 4) {
        const _Float16* Krow = Kbase + (size_t)(jt * 16 + lrow) * PCOLS;
        v8f acc = {};
        acc = __builtin_amdgcn_wmma_f32_16x16x32_f16(false, aq0, false,
                  load_bfrag(Krow, 0, khi), (short)0, acc, false, false);
        acc = __builtin_amdgcn_wmma_f32_16x16x32_f16(false, aq1, false,
                  load_bfrag(Krow, 32, khi), (short)0, acc, false, false);

        bool need_self = row_has_agents && (jt * 16 < AGENT_LIM) &&
                         !((jt * 16 + 15) < band_c0 || (jt * 16) > band_c1);
        v8f accs = {};
        if (need_self) {
            const _Float16* Ksrow = Ksbase + (size_t)(jt * 16 + lrow) * PCOLS;
            accs = __builtin_amdgcn_wmma_f32_16x16x32_f16(false, as0, false,
                       load_bfrag(Ksrow, 0, khi), (short)0, accs, false, false);
            accs = __builtin_amdgcn_wmma_f32_16x16x32_f16(false, as1, false,
                       load_bfrag(Ksrow, 32, khi), (short)0, accs, false, false);
        }

#pragma unroll
        for (int r = 0; r < 8; ++r) {
            int m  = r + khi * 8;               // row within tile
            int ig = it * 16 + m;               // global row
            int jg = jt * 16 + lrow;            // global col
            float s = acc[r] * SCALE;
            if (need_self && ig < AGENT_LIM && jg < AGENT_LIM && (ig / 10) == (jg / 10))
                s = accs[r] * SCALE;
            if (jg >= NN) s = -30000.0f;        // mask padded cols (finite -> no NaN)
            panel[m * NP + jg] = (_Float16)s;
        }
    }
    __syncthreads();

    // ---- phase 2: softmax per row (4 rows per wave, v4h LDS traffic) ----
#pragma unroll
    for (int rr = 0; rr < 4; ++rr) {
        int m = wave * 4 + rr;
        v4h* prow = (v4h*)(panel + m * NP);     // 504 chunks of 4

        float mx = -3.0e38f;
        for (int c = lane; c < NP / 4; c += 32) {
            v4h q = prow[c];
            mx = fmaxf(mx, fmaxf(fmaxf((float)q[0], (float)q[1]),
                                 fmaxf((float)q[2], (float)q[3])));
        }
#pragma unroll
        for (int off = 16; off > 0; off >>= 1)
            mx = fmaxf(mx, __shfl_xor(mx, off, 32));

        float sum = 0.0f;
        for (int c = lane; c < NP / 4; c += 32) {
            v4h q = prow[c];
#pragma unroll
            for (int t = 0; t < 4; ++t) {
                float p = __expf((float)q[t] - mx);
                q[t] = (_Float16)p;
                sum += p;
            }
            prow[c] = q;
        }
#pragma unroll
        for (int off = 16; off > 0; off >>= 1)
            sum += __shfl_xor(sum, off, 32);
        float inv = 1.0f / sum;

        int ig = it * 16 + m;
        float* arow = attn_out + (((size_t)b * HEADS + h) * NN + ig) * (size_t)NN;
        for (int c = lane; c < NP / 4; c += 32) {
            v4h q = prow[c];
            f4 p;
#pragma unroll
            for (int t = 0; t < 4; ++t) {
                p[t] = (float)q[t] * inv;
                q[t] = (_Float16)p[t];
            }
            prow[c] = q;
            int j = c * 4;
            if (ig < NN) {
                if (j + 3 < NN) {
                    *(f4u*)(arow + j) = p;      // 4B-aligned vec store (rows at odd*NN)
                } else {
#pragma unroll
                    for (int t = 0; t < 4; ++t)
                        if (j + t < NN) arow[j + t] = p[t];
                }
            }
        }
    }
    __syncthreads();

    // ---- phase 3: out_head = attn @ V (each wave owns one 16-wide d tile) --
    {
        int d0 = wave * 16;
        const _Float16* panrow = panel + lrow * NP;
        const _Float16* Vrow   = Vt + (((size_t)b * HEADS + h) * DH + d0 + lrow) * NP;
        v8f acc = {};
#pragma unroll 3
        for (int kt = 0; kt < NT / 2; ++kt) {   // 63 k-steps of 32
            int k0 = kt * 32;
            v16h a  = load_afrag(panrow, k0, khi);   // ds_load_b128 x2
            v16h bb = load_bfrag(Vrow, k0, khi);     // global 32B
            acc = __builtin_amdgcn_wmma_f32_16x16x32_f16(false, a, false, bb,
                                                         (short)0, acc, false, false);
        }
#pragma unroll
        for (int r = 0; r < 8; ++r) {
            int row = it * 16 + r + khi * 8;
            O[((size_t)b * NP + row) * DIMM + h * DH + d0 + lrow] = (_Float16)acc[r];
        }
    }
}

// ---------------- kernel 4: output projection + bias (f32 store) ------------
__global__ void __launch_bounds__(256)
k_aaa_outproj(const _Float16* __restrict__ O, const _Float16* __restrict__ wohT,
              const float* __restrict__ b_out, float* __restrict__ out) {
    int wave = (int)((blockIdx.x * 256 + threadIdx.x) >> 5);
    int lane = threadIdx.x & 31;
    const int CT = DIMM / 16;                  // 32
    int ct = wave % CT;
    int rest = wave / CT;
    int mt = rest % NT;
    int b  = rest / NT;
    if (b >= B2) return;

    int lrow = lane & 15;
    int khi  = (lane >= 16) ? 1 : 0;
    const _Float16* Arow = O + ((size_t)b * NP + (size_t)mt * 16 + lrow) * DIMM;
    const _Float16* Brow = wohT + ((size_t)ct * 16 + lrow) * DIMM;

    v8f acc = {};
#pragma unroll 4
    for (int k0 = 0; k0 < DIMM; k0 += 32) {
        v16h a  = load_afrag(Arow, k0, khi);
        v16h bb = load_bfrag(Brow, k0, khi);
        acc = __builtin_amdgcn_wmma_f32_16x16x32_f16(false, a, false, bb,
                                                     (short)0, acc, false, false);
    }
    int col = ct * 16 + lrow;
    float bias = b_out[col];
#pragma unroll
    for (int r = 0; r < 8; ++r) {
        int n = mt * 16 + r + khi * 8;
        if (n < NN)
            out[((size_t)b * NN + n) * DIMM + col] = acc[r] + bias;
    }
}

// ---------------- launcher --------------------------------------------------
extern "C" void kernel_launch(void* const* d_in, const int* in_sizes, int n_in,
                              void* d_out, int out_size, void* d_ws, size_t ws_size,
                              hipStream_t stream) {
    const float* x      = (const float*)d_in[0];
    const float* w_qkv  = (const float*)d_in[1];
    const float* w_self = (const float*)d_in[2];
    const float* w_out  = (const float*)d_in[3];
    const float* b_out  = (const float*)d_in[4];

    float* out      = (float*)d_out;                              // [2, 2009, 512]
    float* attn_out = out + (size_t)B2 * NN * DIMM;               // [2, 8, 2009, 2009]

    // workspace carve-up (f16), ~36 MB total; all section offsets 32B-aligned
    _Float16* xh   = (_Float16*)d_ws;
    _Float16* whT  = xh   + (size_t)B2 * NP * DIMM;               // [PCOLS][DIMM]
    _Float16* wohT = whT  + (size_t)PCOLS * DIMM;                 // [DIMM][DIMM]
    _Float16* P    = wohT + (size_t)DIMM * DIMM;                  // [B][NP][PCOLS]
    _Float16* Vt   = P    + (size_t)B2 * NP * PCOLS;              // [B][H][DH][NP]
    _Float16* O    = Vt   + (size_t)B2 * HEADS * DH * NP;         // [B][NP][DIMM]

    // 1) stage f16 copies (x padded+vectorized, weights transposed)
    {
        size_t nmax = (size_t)PCOLS * DIMM;                       // largest section
        int blocks = (int)((nmax + 255) / 256);
        k_aaa_convert<<<blocks, 256, 0, stream>>>(x, w_qkv, w_self, w_out, xh, whT, wohT);
    }
    // 2) fused projections: 2 * 126 * 160 tiles, 8 waves/block
    {
        int waves = B2 * NT * (PCOLS / 16);
        k_aaa_proj<<<(waves + 7) / 8, 256, 0, stream>>>(xh, whT, P, Vt);
    }
    // 3) attention: one block per (b, h, 16-row tile), 4 waves each
    {
        int blocks = B2 * HEADS * NT;                             // 2016
        k_aaa_attn<<<blocks, 128, 0, stream>>>(P, Vt, attn_out, O);
    }
    // 4) output projection
    {
        int waves = B2 * NT * (DIMM / 16);
        k_aaa_outproj<<<(waves + 7) / 8, 256, 0, stream>>>(O, wohT, b_out, out);
    }
}
